// RWKVB_84026740179191
// MI455X (gfx1250) — compile-verified, matrix operand order checked by reading, not run on hardware
//
#include <hip/hip_runtime.h>
#include <hip/hip_bf16.h>

// ---------------------------------------------------------------------------
// RWKVB forward for MI455X (gfx1250). wave32. All per-token linears run
// through V_WMMA_F32_16X16X32_F16 (f16 in, f32 accumulate). Weights are
// pre-packed (zero-padded in K to a multiple of 32) into B-fragment order so
// the GEMM inner loop is branch-free wide loads + one WMMA per K-step.
// Bi-WKV uses a 3-pass chunked scan (NC=64 chunks of L=64).
// Canonical activation layout: NHWC == [B, T=H*W, C], C contiguous.
// Needs ~370 MB of workspace.
// ---------------------------------------------------------------------------

#define ACT_NONE   0
#define ACT_GELU   1
#define ACT_RELUSQ 2
#define ACT_SIGM   3
#define ACT_RELU   4

static constexpr int Bc    = 16;
static constexpr int Hc    = 64;
static constexpr int Wc    = 64;
static constexpr int Tc    = Hc * Wc;       // 4096
static constexpr int Nrow  = Bc * Tc;       // 65536 tokens
static constexpr int NCc   = 64;            // WKV chunks
static constexpr int Lc    = Tc / NCc;      // 64 steps per chunk

typedef __attribute__((ext_vector_type(16))) _Float16 v16h;
typedef __attribute__((ext_vector_type(8)))  float    v8f;

__device__ __forceinline__ float geluf(float x) {
  return 0.5f * x * (1.0f + erff(x * 0.70710678118654752440f));
}
__device__ __forceinline__ float sigmf(float x) { return 1.0f / (1.0f + expf(-x)); }
__device__ __forceinline__ float apply_act(float v, int act) {
  if (act == ACT_GELU)   return geluf(v);
  if (act == ACT_RELUSQ) { v = fmaxf(v, 0.f); return v * v; }
  if (act == ACT_SIGM)   return sigmf(v);
  if (act == ACT_RELU)   return fmaxf(v, 0.f);
  return v;
}
__device__ __forceinline__ int imin(int a, int b) { return a < b ? a : b; }
__device__ __forceinline__ int imax(int a, int b) { return a > b ? a : b; }

// ---------------------------------------------------------------------------
// Weight repack into B-fragment order, K zero-padded to a multiple of 32.
// Layout: [ntile][ktile][lane 0..31][e 0..15] halves; lane holds column
// n = ntile*16 + (lane&15) and K pattern  k = kt*32 + 8*(lane>>4) +
// (e<8 ? e : e+8)   (ISA §7.12.2 16-bit fragment striping).
// src fp32 is [O,I] when srcOI (1x1 conv) else [I,O] (linear).
// ---------------------------------------------------------------------------
__global__ void pack_wfrag_kernel(const float* __restrict__ w, _Float16* __restrict__ o,
                                  int Cin, int Cout, int srcOI) {
  const int KT = (Cin + 31) >> 5;
  const int NT = Cout >> 4;
  const int total = NT * KT * 512;
  int i = blockIdx.x * blockDim.x + threadIdx.x;
  if (i >= total) return;
  int e    = i & 15;
  int lane = (i >> 4) & 31;
  int kt   = (i >> 9) % KT;
  int nt   = i / (KT * 512);
  int n = nt * 16 + (lane & 15);
  int k = kt * 32 + ((lane >> 4) << 3) + (e < 8 ? e : e + 8);
  float v = 0.f;
  if (k < Cin) v = srcOI ? w[(size_t)n * Cin + k] : w[(size_t)k * Cout + n];
  o[i] = (_Float16)v;
}

// ---------------------------------------------------------------------------
// WMMA GEMM: Y[Nrow,Cout] = act(X[Nrow,Cin] @ W + bias) (+ res)
// One 16x16 tile per wave, K-loop step 32 with v_wmma_f32_16x16x32_f16.
// Requirements (all call sites satisfy them): Cout % 16 == 0; Cin % 16 == 0;
// X buffers have >= 16 floats of tail slack (workspace allocs are padded).
// A fragment = two contiguous 8-float runs per lane -> 4x float4 loads;
// the second run may fall in the zero-padded K region -> branch-free select.
// grid = (Nrow/64, Cout/16), block = 128 (4 waves).
// ---------------------------------------------------------------------------
__launch_bounds__(128)
__global__ void gemm_wmma_kernel(const float* __restrict__ X, const _Float16* __restrict__ W,
                                 const float* bias, const float* res, float* Y,
                                 int Cin, int Cout, int act) {
  const int lane  = threadIdx.x & 31;
  const int wid   = threadIdx.x >> 5;
  const int row0  = (blockIdx.x * 4 + wid) * 16;
  const int ntile = blockIdx.y;
  const int mr    = lane & 15;
  const int hi    = lane >> 4;
  const int KT    = (Cin + 31) >> 5;
  const float*    pA = X + (size_t)(row0 + mr) * Cin + hi * 8;
  const _Float16* pB = W + (((size_t)ntile * KT * 32) + lane) * 16;
  v8f acc = {};
  for (int kt = 0; kt < KT; ++kt) {
    const int kb = kt * 32;
    const float* ap = pA + kb;
    float4 r0 = *(const float4*)(ap);
    float4 r1 = *(const float4*)(ap + 4);
    float4 r2 = *(const float4*)(ap + 16);
    float4 r3 = *(const float4*)(ap + 20);
    __builtin_prefetch(ap + 32, 0, 0);
    // second run valid iff its K start is below Cin (K region is zero-padded
    // in W, but keep A finite/exact with a branch-free select)
    const bool ok2 = (kb + hi * 8 + 16) < Cin;
    r2.x = ok2 ? r2.x : 0.f;  r2.y = ok2 ? r2.y : 0.f;
    r2.z = ok2 ? r2.z : 0.f;  r2.w = ok2 ? r2.w : 0.f;
    r3.x = ok2 ? r3.x : 0.f;  r3.y = ok2 ? r3.y : 0.f;
    r3.z = ok2 ? r3.z : 0.f;  r3.w = ok2 ? r3.w : 0.f;
    v16h a;
    a[0]  = (_Float16)r0.x;  a[1]  = (_Float16)r0.y;
    a[2]  = (_Float16)r0.z;  a[3]  = (_Float16)r0.w;
    a[4]  = (_Float16)r1.x;  a[5]  = (_Float16)r1.y;
    a[6]  = (_Float16)r1.z;  a[7]  = (_Float16)r1.w;
    a[8]  = (_Float16)r2.x;  a[9]  = (_Float16)r2.y;
    a[10] = (_Float16)r2.z;  a[11] = (_Float16)r2.w;
    a[12] = (_Float16)r3.x;  a[13] = (_Float16)r3.y;
    a[14] = (_Float16)r3.z;  a[15] = (_Float16)r3.w;
    v16h bm = *(const v16h*)(pB + (size_t)kt * 512);
    acc = __builtin_amdgcn_wmma_f32_16x16x32_f16(false, a, false, bm, (short)0, acc,
                                                 false, false);
  }
  const int colg = ntile * 16 + mr;
  const float bv = bias ? bias[colg] : 0.f;
#pragma unroll
  for (int r = 0; r < 8; ++r) {
    const int rowg = row0 + hi * 8 + r;           // C/D layout: M = vgpr + 8*hi
    float vv = apply_act(acc[r] + bv, act);
    size_t o = (size_t)rowg * Cout + colg;
    if (res) vv += res[o];
    Y[o] = vv;
  }
}

// ---------------------------------------------------------------------------
// LayerNorm over C per token
// ---------------------------------------------------------------------------
__global__ void ln_kernel(const float* __restrict__ X, const float* g, const float* bb,
                          float* Y, int C) {
  int t = blockIdx.x * blockDim.x + threadIdx.x;
  if (t >= Nrow) return;
  const float* x = X + (size_t)t * C;
  float* y = Y + (size_t)t * C;
  float m = 0.f;
  for (int c = 0; c < C; ++c) m += x[c];
  m *= (1.f / C);
  float var = 0.f;
  for (int c = 0; c < C; ++c) { float d = x[c] - m; var += d * d; }
  var *= (1.f / C);
  float r = rsqrtf(var + 1e-5f);
  for (int c = 0; c < C; ++c) y[c] = (x[c] - m) * r * g[c] + bb[c];
}

// ---------------------------------------------------------------------------
// q_shift + lerp-mix. Quarter 0: left nbr, 1: right, 2: up, 3: down (zero pad).
// ---------------------------------------------------------------------------
__global__ void shift_mix_kernel(const float* __restrict__ X, const float* mk,
                                 const float* mv, const float* mr,
                                 float* ok, float* ov, float* orr, int C) {
  size_t i = (size_t)blockIdx.x * blockDim.x + threadIdx.x;
  if (i >= (size_t)Nrow * C) return;
  int c = (int)(i % C);
  size_t bt = i / C;
  int t = (int)(bt & (Tc - 1));
  int q = C >> 2;
  int ts; bool val;
  if (c < q)           { val = (t & (Wc - 1)) != 0;        ts = t - 1;  }
  else if (c < 2 * q)  { val = (t & (Wc - 1)) != (Wc - 1); ts = t + 1;  }
  else if (c < 3 * q)  { val = t >= Wc;                    ts = t - Wc; }
  else                 { val = t < (Tc - Wc);              ts = t + Wc; }
  size_t base = (bt - (size_t)t) * C;
  float xc = X[i];
  float xs = val ? X[base + (size_t)ts * C + c] : 0.f;
  float a;
  a = mk[c]; ok[i] = xc * a + xs * (1.f - a);
  if (mv && ov) { a = mv[c]; ov[i] = xc * a + xs * (1.f - a); }
  a = mr[c]; orr[i] = xc * a + xs * (1.f - a);
}

// ---------------------------------------------------------------------------
// Depthwise 3x3 (zero pad 1) in NHWC
// ---------------------------------------------------------------------------
__global__ void dwconv_kernel(const float* __restrict__ X, const float* __restrict__ w,
                              const float* bias, float* Y, int C, int act) {
  size_t i = (size_t)blockIdx.x * blockDim.x + threadIdx.x;
  if (i >= (size_t)Nrow * C) return;
  int c = (int)(i % C);
  size_t bt = i / C;
  int t = (int)(bt & (Tc - 1));
  int b = (int)(bt >> 12);
  int y = t >> 6, x = t & (Wc - 1);
  float s = bias[c];
#pragma unroll
  for (int ky = 0; ky < 3; ++ky)
#pragma unroll
    for (int kx = 0; kx < 3; ++kx) {
      int yy = y + ky - 1, xx = x + kx - 1;
      if (yy >= 0 && yy < Hc && xx >= 0 && xx < Wc)
        s += w[c * 9 + ky * 3 + kx] *
             X[(((size_t)b << 12) + (yy << 6) + xx) * C + c];
    }
  Y[i] = apply_act(s, act);
}

// ---------------------------------------------------------------------------
// Bi-WKV chunked scan. d = exp(-exp(decay)/T). Chunk aggregates:
//  forward  Af = sum_i d^(L-1-i) x_i,  backward Ab = sum_i d^i x_i
//  carries compose as c' = d^L c + A (both directions).
// ---------------------------------------------------------------------------
__global__ void wkv_pass1_kernel(const float* __restrict__ K, const float* __restrict__ V,
                                 const float* decay, float* AF, float* BF, float* AB,
                                 float* BB, int C) {
  int i = blockIdx.x * blockDim.x + threadIdx.x;
  if (i >= Bc * NCc * C) return;
  int c = i % C, j = (i / C) % NCc, b = i / (C * NCc);
  float d = expf(-expf(decay[c]) * (1.f / (float)Tc));
  int t0 = j * Lc;
  float af = 0.f, bf = 0.f, ab = 0.f, bb = 0.f;
  for (int t = t0; t < t0 + Lc; ++t) {
    size_t o = ((size_t)b * Tc + t) * C + c;
    float ek = expf(K[o]);
    af = d * af + ek * V[o];
    bf = d * bf + ek;
  }
  for (int t = t0 + Lc - 1; t >= t0; --t) {
    size_t o = ((size_t)b * Tc + t) * C + c;
    float ek = expf(K[o]);
    ab = d * ab + ek * V[o];
    bb = d * bb + ek;
  }
  AF[i] = af; BF[i] = bf; AB[i] = ab; BB[i] = bb;
}

__global__ void wkv_pass2_kernel(const float* decay, const float* AF, const float* BF,
                                 const float* AB, const float* BB,
                                 float* CFa, float* CFd, float* CBa, float* CBd, int C) {
  int i = blockIdx.x * blockDim.x + threadIdx.x;
  if (i >= Bc * C) return;
  int c = i % C, b = i / C;
  float dL = expf(-expf(decay[c]) * ((float)Lc / (float)Tc));
  float ca = 0.f, cd = 0.f;
  for (int j = 0; j < NCc; ++j) {
    size_t o = ((size_t)(b * NCc + j)) * C + c;
    CFa[o] = ca; CFd[o] = cd;
    ca = dL * ca + AF[o]; cd = dL * cd + BF[o];
  }
  ca = 0.f; cd = 0.f;
  for (int j = NCc - 1; j >= 0; --j) {
    size_t o = ((size_t)(b * NCc + j)) * C + c;
    CBa[o] = ca; CBd[o] = cd;
    ca = dL * ca + AB[o]; cd = dL * cd + BB[o];
  }
}

__global__ void wkv_pass3_kernel(const float* __restrict__ K, const float* __restrict__ V,
                                 const float* decay, const float* first,
                                 const float* CFa, const float* CFd,
                                 const float* CBa, const float* CBd,
                                 float* NUM, float* DEN, float* OUT, int C) {
  int i = blockIdx.x * blockDim.x + threadIdx.x;
  if (i >= Bc * NCc * C) return;
  int c = i % C, j = (i / C) % NCc, b = i / (C * NCc);
  float d = expf(-expf(decay[c]) * (1.f / (float)Tc));
  size_t co = ((size_t)(b * NCc + j)) * C + c;
  int t0 = j * Lc;
  float fa = CFa[co], fd = CFd[co];
  for (int t = t0; t < t0 + Lc; ++t) {     // exclusive forward state per t
    size_t o = ((size_t)b * Tc + t) * C + c;
    NUM[o] = fa; DEN[o] = fd;
    float ek = expf(K[o]);
    fa = d * fa + ek * V[o]; fd = d * fd + ek;
  }
  float ba = CBa[co], bd = CBd[co];
  float u = first[c];
  for (int t = t0 + Lc - 1; t >= t0; --t) { // exclusive backward + self term
    size_t o = ((size_t)b * Tc + t) * C + c;
    float kk = K[o], vv = V[o];
    float ek = expf(kk), eu = expf(u + kk);
    OUT[o] = (NUM[o] + ba + eu * vv) / (DEN[o] + bd + eu);
    ba = d * ba + ek * vv; bd = d * bd + ek;
  }
}

// ---------------------------------------------------------------------------
// Elementwise helpers
// ---------------------------------------------------------------------------
__global__ void ew_sigmul_kernel(const float* R, const float* Vv, float* O, long long n) {
  long long i = (long long)blockIdx.x * blockDim.x + threadIdx.x;
  if (i < n) O[i] = sigmf(R[i]) * Vv[i];
}
__global__ void ew_add_sigmul_kernel(float* O, const float* R, const float* Vv, long long n) {
  long long i = (long long)blockIdx.x * blockDim.x + threadIdx.x;
  if (i < n) O[i] += sigmf(R[i]) * Vv[i];
}
__global__ void ew_add_kernel(float* O, const float* A, long long n) {
  long long i = (long long)blockIdx.x * blockDim.x + threadIdx.x;
  if (i < n) O[i] += A[i];
}
__global__ void gate_kernel(const float* E, const float* G, float* O, long long n) {
  long long i = (long long)blockIdx.x * blockDim.x + threadIdx.x;
  if (i < n) O[i] = E[i] * sigmf(G[i]);
}
__global__ void concat_kernel(const float* D1, const float* D2, const float* R2,
                              float* O, long long n) {
  long long i = (long long)blockIdx.x * blockDim.x + threadIdx.x;
  if (i >= n) return;
  int c = (int)(i & 63);
  size_t t = (size_t)(i >> 6);
  float v;
  if (c < 32)      v = D1[t * 32 + c];
  else if (c < 48) v = D2[t * 16 + (c - 32)];
  else             v = R2[t * 16 + (c - 48)];
  O[i] = v;
}

// ---------------------------------------------------------------------------
// RCCA pieces
// ---------------------------------------------------------------------------
__global__ void dec_kernel(const float* __restrict__ DW, const float* w, const float* bsc,
                           float* O) {
  int bt = blockIdx.x * blockDim.x + threadIdx.x;
  if (bt >= Nrow) return;
  const float* p = DW + (size_t)bt * 64;
  float s = bsc[0];
  for (int c = 0; c < 64; ++c) s += w[c] * p[c];
  O[bt] = geluf(s);
}
__global__ void cca_stats_kernel(const float* __restrict__ X, float* Y0) {
  int i = blockIdx.x * blockDim.x + threadIdx.x;
  if (i >= Bc * 64) return;
  int c = i & 63, b = i >> 6;
  float s = 0.f, ss = 0.f;
  for (int t = 0; t < Tc; ++t) {
    float v = X[((size_t)b * Tc + t) * 64 + c];
    s += v; ss += v * v;
  }
  float m = s * (1.f / Tc);
  float var = ss * (1.f / Tc) - m * m;
  Y0[i] = sqrtf(fmaxf(var, 0.f)) + m;
}
__global__ void cca_mlp_kernel(const float* Y0, const float* w1, const float* b1,
                               const float* w2, const float* b2, float* SCL) {
  int b = blockIdx.x * blockDim.x + threadIdx.x;
  if (b >= Bc) return;
  float h[4];
  for (int j = 0; j < 4; ++j) {
    float s = b1[j];
    for (int c = 0; c < 64; ++c) s += w1[j * 64 + c] * Y0[b * 64 + c];
    h[j] = fmaxf(s, 0.f);
  }
  for (int c = 0; c < 64; ++c) {
    float s = b2[c];
    for (int j = 0; j < 4; ++j) s += w2[c * 4 + j] * h[j];
    SCL[b * 64 + c] = sigmf(s);
  }
}
__global__ void rcca_combine_kernel(const float* DW, const float* DEC, const float* SCL,
                                    const float* sigma, const float* M, const float* Xin,
                                    float* O, long long n) {
  long long i = (long long)blockIdx.x * blockDim.x + threadIdx.x;
  if (i >= n) return;
  int c = (int)(i & 63);
  size_t bt = (size_t)(i >> 6);
  int b = (int)(bt >> 12);
  float dw = DW[i];
  O[i] = sigma[c] * (dw - DEC[bt]) + dw * SCL[b * 64 + c] + M[i] + Xin[i];
}

// ---------------------------------------------------------------------------
// ESA small convs (NHWC direct)
// ---------------------------------------------------------------------------
__global__ void conv3x3_kernel(const float* __restrict__ X, const float* __restrict__ w,
                               const float* bias, float* Y, int Hi, int Wi, int Ci,
                               int Ho, int Wo, int Co, int stride, int pad, int act) {
  int i = blockIdx.x * blockDim.x + threadIdx.x;
  int total = Bc * Ho * Wo * Co;
  if (i >= total) return;
  int co = i % Co, xo = (i / Co) % Wo, yo = (i / (Co * Wo)) % Ho, b = i / (Co * Wo * Ho);
  float s = bias[co];
  for (int ky = 0; ky < 3; ++ky)
    for (int kx = 0; kx < 3; ++kx) {
      int yy = yo * stride + ky - pad, xx = xo * stride + kx - pad;
      if (yy >= 0 && yy < Hi && xx >= 0 && xx < Wi) {
        const float* xp = X + ((size_t)(b * Hi + yy) * Wi + xx) * Ci;
        const float* wp = w + (size_t)co * Ci * 9 + ky * 3 + kx;
        for (int ci = 0; ci < Ci; ++ci) s += wp[ci * 9] * xp[ci];
      }
    }
  Y[i] = apply_act(s, act);
}
__global__ void maxpool_kernel(const float* __restrict__ X, float* Y) {
  // [B,31,31,16] -> [B,9,9,16], 7x7 window, stride 3, VALID
  int i = blockIdx.x * blockDim.x + threadIdx.x;
  if (i >= Bc * 9 * 9 * 16) return;
  int c = i % 16, xo = (i / 16) % 9, yo = (i / 144) % 9, b = i / 1296;
  float m = -INFINITY;
  for (int ky = 0; ky < 7; ++ky)
    for (int kx = 0; kx < 7; ++kx) {
      int yy = yo * 3 + ky, xx = xo * 3 + kx;
      m = fmaxf(m, X[((size_t)(b * 31 + yy) * 31 + xx) * 16 + c]);
    }
  Y[i] = m;
}
__device__ __forceinline__ float bil_get(const float* X, int b, int yy, int xx, int c) {
  return X[((size_t)(b * 9 + yy) * 9 + xx) * 16 + c];
}
__global__ void resize_kernel(const float* __restrict__ X, float* Y) {
  // bilinear [B,9,9,16] -> [B,64,64,16], half-pixel (align_corners=False)
  int i = blockIdx.x * blockDim.x + threadIdx.x;
  if (i >= Bc * 64 * 64 * 16) return;
  int c = i % 16, xo = (i / 16) & 63, yo = (i / 1024) & 63, b = i >> 16;
  float fy = (yo + 0.5f) * (9.f / 64.f) - 0.5f;
  float fx = (xo + 0.5f) * (9.f / 64.f) - 0.5f;
  int y0 = (int)floorf(fy), x0 = (int)floorf(fx);
  float wy = fy - y0, wx = fx - x0;
  int y0c = imin(imax(y0, 0), 8), y1c = imin(imax(y0 + 1, 0), 8);
  int x0c = imin(imax(x0, 0), 8), x1c = imin(imax(x0 + 1, 0), 8);
  float v = (1 - wy) * ((1 - wx) * bil_get(X, b, y0c, x0c, c) + wx * bil_get(X, b, y0c, x1c, c)) +
            wy       * ((1 - wx) * bil_get(X, b, y1c, x0c, c) + wx * bil_get(X, b, y1c, x1c, c));
  Y[i] = v;
}

// ---------------------------------------------------------------------------
// Host
// ---------------------------------------------------------------------------
static inline dim3 g1(long long n, int bs) { return dim3((unsigned)((n + bs - 1) / bs)); }

extern "C" void kernel_launch(void* const* d_in, const int* in_sizes, int n_in,
                              void* d_out, int out_size, void* d_ws, size_t ws_size,
                              hipStream_t stream) {
  (void)in_sizes; (void)n_in; (void)out_size; (void)ws_size;

  // ---- input unpack: depth-first, dict-insertion order (108 leaves) ----
  int ii = 0;
  auto F32 = [&]() { return (const float*)d_in[ii++]; };
  struct MB  { const float *w1, *b1, *wd, *bd, *w2, *b2; };
  struct BLK { const float *ln1g, *ln1b, *ln2g, *ln2b,
                           *smk, *smv, *smr, *Wk, *Wv, *Wr, *Wo, *decay, *first,
                           *cmk, *cmr, *cWk, *cWv, *cWr; };
  auto rdMB = [&]() { MB m; m.w1=F32(); m.b1=F32(); m.wd=F32(); m.bd=F32(); m.w2=F32(); m.b2=F32(); return m; };
  auto rdBLK = [&]() { BLK p;
    p.ln1g=F32(); p.ln1b=F32(); p.ln2g=F32(); p.ln2b=F32();
    p.smk=F32(); p.smv=F32(); p.smr=F32();
    p.Wk=F32(); p.Wv=F32(); p.Wr=F32(); p.Wo=F32(); p.decay=F32(); p.first=F32();
    p.cmk=F32(); p.cmr=F32(); p.cWk=F32(); p.cWv=F32(); p.cWr=F32(); return p; };

  const float* x_in = F32();
  ii += 2;                                   // H, W (hardcoded 64)
  MB  conv1 = rdMB();  BLK blk1 = rdBLK();
  const float *d1w = F32(), *d1b = F32(), *r1w = F32(), *r1b = F32();
  MB  conv2 = rdMB();  BLK blk2 = rdBLK();
  const float *d2w = F32(), *d2b = F32(), *r2w = F32(), *r2b = F32();
  MB  conv3 = rdMB();  BLK blk3 = rdBLK();
  const float *mixw = F32(), *mixb = F32();
  const float *rc_dww = F32(), *rc_dwb = F32(), *rc_decw = F32(), *rc_decb = F32(),
              *rc_sig = F32(), *rc_cw1 = F32(), *rc_cb1 = F32(), *rc_cw2 = F32(),
              *rc_cb2 = F32();
  const float *e_w1 = F32(), *e_b1 = F32(), *e_wf = F32(), *e_bf = F32(),
              *e_wmax = F32(), *e_bmax = F32(), *e_w2 = F32(), *e_b2 = F32(),
              *e_w3 = F32(), *e_b3 = F32(), *e_w3_ = F32(), *e_b3_ = F32(),
              *e_w4 = F32(), *e_b4 = F32();

  // ---- workspace carve (every buffer gets +64 floats of tail slack for the
  //      GEMM's padded-K overread) ----
  char* ws = (char*)d_ws;
  size_t off = 0;
  auto alloc_f = [&](size_t nflt) {
    float* p = (float*)(ws + off);
    off += (((nflt + 64) * sizeof(float)) + 255) & ~(size_t)255;
    return p;
  };
  const size_t NF = (size_t)Nrow * 64;
  float* WIDE0 = alloc_f((size_t)Nrow * 256);   // mbconv dw out / cm hidden
  float* WIDE1 = alloc_f((size_t)Nrow * 128);   // mbconv pw1 (expand) out
  float* Fb  = alloc_f(NF);  float* Ab  = alloc_f(NF);  float* Lb  = alloc_f(NF);
  float* XKb = alloc_f(NF);  float* XVb = alloc_f(NF);  float* XRb = alloc_f(NF);
  float* Kb  = alloc_f(NF);  float* Vb  = alloc_f(NF);  float* Rb  = alloc_f(NF);
  float* D1b = alloc_f((size_t)Nrow * 32);
  float* D2b = alloc_f((size_t)Nrow * 16);
  float* CCb = alloc_f(NF);  float* Mxb = alloc_f(NF);  float* DWb = alloc_f(NF);
  float* Eb  = alloc_f(NF);  float* C4b = alloc_f(NF);
  float* DECb = alloc_f((size_t)Nrow);
  float* Y0b  = alloc_f(Bc * 64);  float* SCLb = alloc_f(Bc * 64);
  float* C1_b = alloc_f((size_t)Nrow * 16);
  float* C1b  = alloc_f((size_t)Bc * 31 * 31 * 16);
  float* VMb  = alloc_f((size_t)Bc * 81 * 16);
  float* VRb  = alloc_f((size_t)Bc * 81 * 16);
  float* C3A  = alloc_f((size_t)Bc * 81 * 16);
  float* C3B  = alloc_f((size_t)Bc * 81 * 16);
  float* C3R  = alloc_f((size_t)Nrow * 16);
  float* CFb2 = alloc_f((size_t)Nrow * 16);
  const size_t CH = (size_t)Bc * NCc * 64;
  float *AF=alloc_f(CH), *BF=alloc_f(CH), *AB=alloc_f(CH), *BB=alloc_f(CH);
  float *CFa=alloc_f(CH), *CFd=alloc_f(CH), *CBa=alloc_f(CH), *CBd=alloc_f(CH);
  _Float16* wpack = (_Float16*)(ws + off);  off += 4u * 1024 * 1024;
  size_t woff = 0;

  // ---- launch helpers ----
  auto pack = [&](const float* w, int Cin, int Cout, bool srcOI) {
    const int KT = (Cin + 31) >> 5;
    const int NT = Cout >> 4;
    _Float16* dst = wpack + woff;
    const int n = NT * KT * 512;
    woff += (size_t)n;
    pack_wfrag_kernel<<<g1(n, 256), 256, 0, stream>>>(w, dst, Cin, Cout, srcOI ? 1 : 0);
    return (const _Float16*)dst;
  };
  auto gemm = [&](const float* X, const _Float16* W, const float* bias, const float* res,
                  float* Y, int Cin, int Cout, int act) {
    dim3 grd(Nrow / 64, Cout / 16);           // Cout % 16 == 0 at every call site
    gemm_wmma_kernel<<<grd, 128, 0, stream>>>(X, W, bias, res, Y, Cin, Cout, act);
  };
  auto mbconv = [&](const float* in, const MB& m, int d, float* outF) {
    const _Float16* W1 = pack(m.w1, d, 2 * d, true);
    const _Float16* W2 = pack(m.w2, 2 * d, d, true);
    gemm(in, W1, m.b1, nullptr, WIDE1, d, 2 * d, ACT_GELU);
    dwconv_kernel<<<g1((long long)Nrow * 2 * d, 256), 256, 0, stream>>>(
        WIDE1, m.wd, m.bd, WIDE0, 2 * d, ACT_GELU);
    gemm(WIDE0, W2, m.b2, nullptr, outF, 2 * d, d, ACT_NONE);
  };
  auto rwkv_block = [&](float* Fin, float* Aout, const BLK& p, int C) {
    const _Float16* Wk  = pack(p.Wk,  C, C, false);
    const _Float16* Wv  = pack(p.Wv,  C, C, false);
    const _Float16* Wr  = pack(p.Wr,  C, C, false);
    const _Float16* Wo  = pack(p.Wo,  C, C, false);
    const _Float16* cWk = pack(p.cWk, C, 4 * C, false);
    const _Float16* cWv = pack(p.cWv, 4 * C, C, false);
    const _Float16* cWr = pack(p.cWr, C, C, false);
    long long nC = (long long)Nrow * C;
    // spatial mix
    ln_kernel<<<g1(Nrow, 256), 256, 0, stream>>>(Fin, p.ln1g, p.ln1b, Lb, C);
    shift_mix_kernel<<<g1(nC, 256), 256, 0, stream>>>(Lb, p.smk, p.smv, p.smr,
                                                      XKb, XVb, XRb, C);
    gemm(XKb, Wk, nullptr, nullptr, Kb, C, C, ACT_NONE);
    gemm(XVb, Wv, nullptr, nullptr, Vb, C, C, ACT_NONE);
    gemm(XRb, Wr, nullptr, nullptr, Rb, C, C, ACT_NONE);
    // bi-WKV (NUM/DEN overlay dead XKb/XVb, OUT overlays dead XRb)
    wkv_pass1_kernel<<<g1(Bc * NCc * C, 256), 256, 0, stream>>>(Kb, Vb, p.decay,
                                                                AF, BF, AB, BB, C);
    wkv_pass2_kernel<<<g1(Bc * C, 256), 256, 0, stream>>>(p.decay, AF, BF, AB, BB,
                                                          CFa, CFd, CBa, CBd, C);
    wkv_pass3_kernel<<<g1(Bc * NCc * C, 256), 256, 0, stream>>>(
        Kb, Vb, p.decay, p.first, CFa, CFd, CBa, CBd, XKb, XVb, XRb, C);
    ew_sigmul_kernel<<<g1(nC, 256), 256, 0, stream>>>(Rb, XRb, XKb, nC); // G=sig(r)*wkv
    gemm(XKb, Wo, nullptr, Fin, Aout, C, C, ACT_NONE);                   // A = F + sm
    // channel mix
    ln_kernel<<<g1(Nrow, 256), 256, 0, stream>>>(Aout, p.ln2g, p.ln2b, Lb, C);
    shift_mix_kernel<<<g1(nC, 256), 256, 0, stream>>>(Lb, p.cmk, nullptr, p.cmr,
                                                      XKb, nullptr, XRb, C);
    gemm(XKb, cWk, nullptr, nullptr, WIDE0, C, 4 * C, ACT_RELUSQ);
    gemm(XRb, cWr, nullptr, nullptr, Rb, C, C, ACT_NONE);
    gemm(WIDE0, cWv, nullptr, nullptr, Vb, 4 * C, C, ACT_NONE);
    ew_add_sigmul_kernel<<<g1(nC, 256), 256, 0, stream>>>(Aout, Rb, Vb, nC);
    ew_add_kernel<<<g1(nC, 256), 256, 0, stream>>>(Aout, Fin, nC);       // outer shortcut
  };

  // ---- stage 1 (C=64) ----
  mbconv(x_in, conv1, 64, Fb);
  rwkv_block(Fb, Ab, blk1, 64);
  gemm(Ab, pack(d1w, 64, 32, true), d1b, nullptr, D1b, 64, 32, ACT_NONE);
  gemm(Ab, pack(r1w, 64, 32, true), r1b, nullptr, Lb, 64, 32, ACT_NONE);
  // ---- stage 2 (C=32) ----
  mbconv(Lb, conv2, 32, Fb);
  rwkv_block(Fb, Ab, blk2, 32);
  gemm(Ab, pack(d2w, 32, 16, true), d2b, nullptr, D2b, 32, 16, ACT_NONE);
  gemm(Ab, pack(r2w, 32, 16, true), r2b, nullptr, Lb, 32, 16, ACT_NONE);
  // ---- stage 3 (C=16) ----
  mbconv(Lb, conv3, 16, Fb);
  rwkv_block(Fb, Ab, blk3, 16);

  // ---- concat + mixer ----
  long long n64 = (long long)Nrow * 64;
  concat_kernel<<<g1(n64, 256), 256, 0, stream>>>(D1b, D2b, Ab, CCb, n64);
  gemm(CCb, pack(mixw, 64, 64, true), mixb, nullptr, Mxb, 64, 64, ACT_NONE);

  // ---- RCCA (+ outer shortcut x_in fused) ----
  dwconv_kernel<<<g1(n64, 256), 256, 0, stream>>>(Mxb, rc_dww, rc_dwb, DWb, 64, ACT_GELU);
  dec_kernel<<<g1(Nrow, 256), 256, 0, stream>>>(DWb, rc_decw, rc_decb, DECb);
  cca_stats_kernel<<<g1(Bc * 64, 256), 256, 0, stream>>>(DWb, Y0b);
  cca_mlp_kernel<<<1, 32, 0, stream>>>(Y0b, rc_cw1, rc_cb1, rc_cw2, rc_cb2, SCLb);
  rcca_combine_kernel<<<g1(n64, 256), 256, 0, stream>>>(DWb, DECb, SCLb, rc_sig,
                                                        Mxb, x_in, Eb, n64);
  // ---- ESA ----
  gemm(Eb, pack(e_w1, 64, 16, true), e_b1, nullptr, C1_b, 64, 16, ACT_NONE);
  conv3x3_kernel<<<g1((long long)Bc * 31 * 31 * 16, 256), 256, 0, stream>>>(
      C1_b, e_w2, e_b2, C1b, 64, 64, 16, 31, 31, 16, 2, 0, ACT_NONE);
  maxpool_kernel<<<g1((long long)Bc * 81 * 16, 256), 256, 0, stream>>>(C1b, VMb);
  conv3x3_kernel<<<g1((long long)Bc * 81 * 16, 256), 256, 0, stream>>>(
      VMb, e_wmax, e_bmax, VRb, 9, 9, 16, 9, 9, 16, 1, 1, ACT_RELU);
  conv3x3_kernel<<<g1((long long)Bc * 81 * 16, 256), 256, 0, stream>>>(
      VRb, e_w3, e_b3, C3A, 9, 9, 16, 9, 9, 16, 1, 1, ACT_RELU);
  conv3x3_kernel<<<g1((long long)Bc * 81 * 16, 256), 256, 0, stream>>>(
      C3A, e_w3_, e_b3_, C3B, 9, 9, 16, 9, 9, 16, 1, 1, ACT_NONE);
  resize_kernel<<<g1((long long)Nrow * 16, 256), 256, 0, stream>>>(C3B, C3R);
  gemm(C1_b, pack(e_wf, 16, 16, true), e_bf, nullptr, CFb2, 16, 16, ACT_NONE);
  ew_add_kernel<<<g1((long long)Nrow * 16, 256), 256, 0, stream>>>(C3R, CFb2,
                                                                   (long long)Nrow * 16);
  gemm(C3R, pack(e_w4, 16, 64, true), e_b4, nullptr, C4b, 16, 64, ACT_NONE);
  gate_kernel<<<g1(n64, 256), 256, 0, stream>>>(Eb, C4b, (float*)d_out, n64);
}
